// CrossAttnBlock_54589034332171
// MI455X (gfx1250) — compile-verified, matrix operand order checked by reading, not run on hardware
//
#include <hip/hip_runtime.h>
#include <hip/hip_bf16.h>
#include <math.h>

// ---------------------------------------------------------------------------
// CrossAttnBlock (Swin cross-window attention + MLP) for gfx1250 / MI455X.
// All GEMMs run on v_wmma_f32_16x16x32_f16 (f16 in, f32 accumulate).
// B tiles are staged into LDS via the Tensor Data Mover (tensor_load_to_lds,
// TENSORcnt-synchronized, double-buffered) when the builtin is available.
// ---------------------------------------------------------------------------

typedef __attribute__((ext_vector_type(16))) _Float16 v16h;
typedef __attribute__((ext_vector_type(8)))  _Float16 v8h;
typedef __attribute__((ext_vector_type(8)))  float    v8f;
typedef unsigned int u32x4 __attribute__((ext_vector_type(4)));
typedef int          i32x4 __attribute__((ext_vector_type(4)));
typedef int          i32x8 __attribute__((ext_vector_type(8)));

#if defined(__has_builtin)
#  if __has_builtin(__builtin_amdgcn_tensor_load_to_lds)
#    define HAVE_TDM 1
#  else
#    define HAVE_TDM 0
#  endif
#else
#  define HAVE_TDM 0
#endif

#define HH   64
#define WW   64
#define CC   256
#define NHD  8
#define HD   32
#define WS   8
#define NTK  64           // tokens per window
#define ROWS 65536        // B * H * W = 16 * 4096
#define ATTN_SCALE 0.17677669529663687f   // 32^-0.5

// ---------------- WMMA helpers ----------------------------------------------

__device__ __forceinline__ v8f wmma_f16(v16h a, v16h b, v8f c) {
  // D = A(16x32 f16) * B(32x16 f16) + C(16x16 f32)
  return __builtin_amdgcn_wmma_f32_16x16x32_f16(
      /*neg_a=*/false, a, /*neg_b=*/false, b,
      /*c_mod=*/(short)0, c, /*reuse_a=*/false, /*reuse_b=*/false);
}

// Load a 16x32 f16 fragment (A layout; also B when source is stored N-major,
// i.e. row = n, contiguous in K).  Per ISA 7.12.2: lanes 0..15 -> rows, half
// wave selects which 8-wide K group; groups are contiguous -> two b128 loads.
__device__ __forceinline__ v16h load_frag(const _Float16* base, int row,
                                          int k0, int ld, int lane) {
  int m  = lane & 15;
  int hh = lane >> 4;                 // 0 or 1
  const _Float16* p = base + (size_t)(row + m) * ld + k0 + hh * 8;
  v8h lo = *(const v8h*)(p);          // K = k0 + hh*8      .. +7
  v8h hi = *(const v8h*)(p + 16);     // K = k0 + 16 + hh*8 .. +7
  v16h r;
#pragma unroll
  for (int i = 0; i < 8; ++i) { r[i] = lo[i]; r[i + 8] = hi[i]; }
  return r;
}

// ---------------- B-tile staging: global (N-major) -> LDS -------------------
// Stages a 64-row x 32-col f16 tile of Bt (leading dim K) into dstLds.
// TDM path: single wave issues a 2D tensor_load_to_lds (D# built in SGPRs);
// fallback: cooperative 256-thread vector copy.

__device__ __forceinline__ void stage_B(const _Float16* __restrict__ Bt, int n0,
                                        int K, int ks, _Float16* dstLds, int tid) {
#if HAVE_TDM
  if ((tid >> 5) == 0) {   // one wave issues the DMA (TDM ignores EXEC)
    unsigned long long ga =
        (unsigned long long)(uintptr_t)(Bt + (size_t)n0 * K + (size_t)ks * 32);
    unsigned lds = (unsigned)(uintptr_t)(void*)dstLds;   // low 32b = LDS offset
    u32x4 g0;
    g0[0] = 1u;                                          // count = 1 descriptor
    g0[1] = lds;                                         // lds_addr
    g0[2] = (unsigned)ga;                                // global_addr[31:0]
    g0[3] = ((unsigned)(ga >> 32) & 0x1FFFFFFu) | (2u << 30);  // ga[56:32]|type=2
    i32x8 g1;
    g1[0] = 0x00010000;                                  // data_size = 2 bytes
    g1[1] = (int)(((unsigned)K & 0xFFFFu) << 16);        // tensor_dim0 lo16
    g1[2] = (int)((((unsigned)K >> 16) & 0xFFFFu) | (64u << 16)); // dim0 hi | dim1 lo
    g1[3] = (int)(32u << 16);                            // dim1 hi=0 | tile_dim0=32
    g1[4] = 64;                                          // tile_dim1=64, tile_dim2=0
    g1[5] = K;                                           // tensor_dim0_stride lo32
    g1[6] = 0;
    g1[7] = 0;
    i32x4 z4 = {0, 0, 0, 0};
#if __clang_major__ >= 23
    i32x8 z8 = {0, 0, 0, 0, 0, 0, 0, 0};
    __builtin_amdgcn_tensor_load_to_lds(g0, g1, z4, z4, z8, 0);
#else
    __builtin_amdgcn_tensor_load_to_lds(g0, g1, z4, z4, 0);
#endif
  }
#else
  int r = tid >> 2, c = (tid & 3) * 8;                   // 64 rows x 4 v8h chunks
  *(v8h*)(dstLds + r * 32 + c) =
      *(const v8h*)(Bt + (size_t)(n0 + r) * K + (size_t)ks * 32 + c);
#endif
}

// ---------------- index remaps ----------------------------------------------

__device__ __forceinline__ int remap_i2w(int row) {   // image row -> window row
  int b = row >> 12, r = row & 4095;
  int y = r >> 6, x = r & 63;
  return (b << 12) + (((y >> 3) * 8 + (x >> 3)) << 6) + ((y & 7) << 3) + (x & 7);
}
__device__ __forceinline__ int remap_w2i(int m) {     // window row -> image row
  int b = m >> 12, r = m & 4095;
  int win = r >> 6, idx = r & 63;
  int y = (win >> 3) * 8 + (idx >> 3);
  int x = (win & 7) * 8 + (idx & 7);
  return (b << 12) + y * 64 + x;
}

// ---------------- weight prep: f32 KxN -> f16 NxK ---------------------------

__global__ void wprep_kernel(const float* __restrict__ W, _Float16* __restrict__ Wt,
                             int K, int Nn) {
  int idx = blockIdx.x * 256 + threadIdx.x;
  if (idx < K * Nn) {
    int k = idx / Nn, n = idx - k * Nn;
    Wt[(size_t)n * K + k] = (_Float16)W[idx];
  }
}

// ---------------- LayerNorm (optionally fused window partition) -------------

__global__ __launch_bounds__(64)
void ln_kernel(const float* __restrict__ x, const float* __restrict__ g,
               const float* __restrict__ b, _Float16* __restrict__ out,
               int partition) {
  int row = blockIdx.x;
  int tid = threadIdx.x;                       // 64 threads, 4 ch each
  const float4 v4 = ((const float4*)(x + (size_t)row * CC))[tid];
  float s1 = v4.x + v4.y + v4.z + v4.w;
  float s2 = v4.x * v4.x + v4.y * v4.y + v4.z * v4.z + v4.w * v4.w;
  __shared__ float r1[64], r2[64];
  r1[tid] = s1; r2[tid] = s2;
  __syncthreads();
#pragma unroll
  for (int off = 32; off > 0; off >>= 1) {
    if (tid < off) { r1[tid] += r1[tid + off]; r2[tid] += r2[tid + off]; }
    __syncthreads();
  }
  float mean = r1[0] * (1.0f / CC);
  float var  = r2[0] * (1.0f / CC) - mean * mean;
  float rstd = rsqrtf(var + 1e-5f);
  int orow = partition ? remap_i2w(row) : row;
  _Float16* o = out + (size_t)orow * CC + tid * 4;
  const float* gg = g + tid * 4; const float* bb = b + tid * 4;
  float v[4] = {v4.x, v4.y, v4.z, v4.w};
#pragma unroll
  for (int i = 0; i < 4; ++i)
    o[i] = (_Float16)((v[i] - mean) * rstd * gg[i] + bb[i]);
}

// ---------------- generic WMMA GEMM -----------------------------------------
// A: MxK f16 row-major.  Bt: NxK f16 (pre-transposed weight).  128x64 tile,
// 8 waves, each wave 16 rows x 64 cols (4 accumulators, A reused 4x).
// B tile for each 32-wide K step staged in LDS (TDM double-buffered).
// OUTMODE 0: f16 store row-major            (+optional GELU via ACT)
//         1: f32 store, out = resid + acc   (residual add)
//         2: f32 store with window->image row remap, out = resid + acc
//         3: f16 scatter to per-head Q layout   (w,h,t,d)
//         4: f16 scatter, cols [0,256)->K buf, [256,512)->V buf

template <int ACT, int OUTMODE>
__global__ __launch_bounds__(256)
void gemm_wmma(const _Float16* __restrict__ A, const _Float16* __restrict__ Bt,
               const float* __restrict__ bias, void* __restrict__ out0,
               void* __restrict__ out1, const float* __restrict__ resid,
               int M, int Nn, int K) {
  __shared__ _Float16 Bs[2][64 * 32];          // double-buffered B tile (8 KB)
  int tid  = threadIdx.x;
  int wave = tid >> 5;
  int lane = tid & 31;
  int m0 = blockIdx.x * 128 + wave * 16;
  int n0 = blockIdx.y * 64;
  int nk = K >> 5;

  stage_B(Bt, n0, K, 0, &Bs[0][0], tid);       // prologue DMA

  v8f acc[4] = {v8f{}, v8f{}, v8f{}, v8f{}};
  for (int ks = 0; ks < nk; ++ks) {
    int buf = ks & 1;
#if HAVE_TDM
    if (wave == 0) __builtin_amdgcn_s_wait_tensorcnt((short)0);
#endif
    __syncthreads();                           // Bs[buf] ready for all waves
    if (ks + 1 < nk)                           // overlap next DMA with WMMAs
      stage_B(Bt, n0, K, ks + 1, &Bs[buf ^ 1][0], tid);
    if (ks + 2 < nk)                           // prefetch streaming A operand
      __builtin_prefetch(A + (size_t)(m0 + (lane & 15)) * K + (ks + 2) * 32, 0, 1);

    v16h a = load_frag(A, m0, ks * 32, K, lane);
#pragma unroll
    for (int t = 0; t < 4; ++t) {
      v16h b = load_frag(&Bs[buf][0], t * 16, 0, 32, lane);   // ds_load_b128
      acc[t] = wmma_f16(a, b, acc[t]);
    }
  }

  int col = lane & 15;
  int rh  = (lane >> 4) * 8;
#pragma unroll
  for (int t = 0; t < 4; ++t) {
    int n = n0 + t * 16 + col;
    float bv = bias[n];
#pragma unroll
    for (int r = 0; r < 8; ++r) {
      int m = m0 + rh + r;
      float v = acc[t][r] + bv;
      if (ACT == 1) v = 0.5f * v * (1.0f + erff(v * 0.70710678118654752f));
      if (OUTMODE == 0) {
        ((_Float16*)out0)[(size_t)m * Nn + n] = (_Float16)v;
      } else if (OUTMODE == 1) {
        size_t i = (size_t)m * Nn + n;
        ((float*)out0)[i] = resid[i] + v;
      } else if (OUTMODE == 2) {
        size_t i = (size_t)remap_w2i(m) * Nn + n;
        ((float*)out0)[i] = resid[i] + v;
      } else if (OUTMODE == 3) {
        int w = m >> 6, tok = m & 63, h = n >> 5, d = n & 31;
        ((_Float16*)out0)[(((size_t)w * NHD + h) * NTK + tok) * HD + d] = (_Float16)v;
      } else {                                    // OUTMODE == 4
        int c = n & 255;
        _Float16* dst = (n < 256) ? (_Float16*)out0 : (_Float16*)out1;
        int w = m >> 6, tok = m & 63, h = c >> 5, d = c & 31;
        dst[(((size_t)w * NHD + h) * NTK + tok) * HD + d] = (_Float16)v;
      }
    }
  }
}

// ---------------- attention: one (window, head) per block -------------------
// S = (q*scale) @ k^T + rel_bias ; P = softmax(S) ; O = P @ v
// q,k,v per head: 64x32 f16 contiguous.  S staged in LDS, P and V^T re-read
// from LDS as WMMA fragments (ds_load_b128 path).

__global__ __launch_bounds__(128)
void attn_kernel(const _Float16* __restrict__ qh, const _Float16* __restrict__ kh,
                 const _Float16* __restrict__ vh, const float* __restrict__ rel,
                 _Float16* __restrict__ outw) {
  int wh   = blockIdx.x;              // window * NHD + head
  int head = wh & (NHD - 1);
  int w    = wh >> 3;
  const _Float16* q = qh + (size_t)wh * NTK * HD;
  const _Float16* k = kh + (size_t)wh * NTK * HD;
  const _Float16* v = vh + (size_t)wh * NTK * HD;

  __shared__ float    S[64][68];      // padded to dodge bank conflicts
  __shared__ _Float16 P[64][64];      // 16B-aligned rows for b128 frag loads
  __shared__ _Float16 VT[32][64];     // v transposed (d-major) for B frags

  int tid = threadIdx.x, wave = tid >> 5, lane = tid & 31;

  // stage V^T into LDS
  for (int i = tid; i < NTK * HD; i += 128) {
    int t = i >> 5, d = i & 31;
    VT[d][t] = v[i];
  }

  // --- S = q @ k^T  (K = 32, one wmma per 16x16 tile) ---
  v8f s[4] = {v8f{}, v8f{}, v8f{}, v8f{}};
  {
    v16h a = load_frag(q, wave * 16, 0, HD, lane);
#pragma unroll
    for (int t = 0; t < 4; ++t) {
      v16h b = load_frag(k, t * 16, 0, HD, lane);   // k rows are "N-major"
      s[t] = wmma_f16(a, b, s[t]);
    }
  }
  int col = lane & 15, rh = (lane >> 4) * 8;
#pragma unroll
  for (int t = 0; t < 4; ++t)
#pragma unroll
    for (int r = 0; r < 8; ++r)
      S[wave * 16 + rh + r][t * 16 + col] = s[t][r] * ATTN_SCALE;
  __syncthreads();

  // --- row softmax with Swin relative-position bias (computed in place) ---
  if (tid < 64) {
    int i = tid, yi = i >> 3, xi = i & 7;
    float mx = -3.0e38f;
    for (int j = 0; j < 64; ++j) {
      int yj = j >> 3, xj = j & 7;
      int ridx = (yi - yj + WS - 1) * (2 * WS - 1) + (xi - xj + WS - 1);
      float val = S[i][j] + rel[ridx * NHD + head];
      S[i][j] = val;
      mx = fmaxf(mx, val);
    }
    float sum = 0.0f;
    for (int j = 0; j < 64; ++j) { float e = expf(S[i][j] - mx); sum += e; S[i][j] = e; }
    float inv = 1.0f / sum;
    for (int j = 0; j < 64; ++j) P[i][j] = (_Float16)(S[i][j] * inv);
  }
  __syncthreads();

  // --- O = P @ V : M=16/wave, K=64 (two steps), N=32 (two tiles) ---
  v8f o[2] = {v8f{}, v8f{}};
#pragma unroll
  for (int k0 = 0; k0 < 64; k0 += 32) {
    v16h a = load_frag(&P[0][0], wave * 16, k0, 64, lane);
#pragma unroll
    for (int t = 0; t < 2; ++t) {
      v16h b = load_frag(&VT[0][0], t * 16, k0, 64, lane);
      o[t] = wmma_f16(a, b, o[t]);
    }
  }
  // write to window-layout token matrix (B_, N, C) at column head*32
#pragma unroll
  for (int t = 0; t < 2; ++t)
#pragma unroll
    for (int r = 0; r < 8; ++r) {
      int m = wave * 16 + rh + r;
      int n = head * HD + t * 16 + col;
      outw[((size_t)w * NTK + m) * CC + n] = (_Float16)o[t][r];
    }
}

// ---------------------------------------------------------------------------

extern "C" void kernel_launch(void* const* d_in, const int* in_sizes, int n_in,
                              void* d_out, int out_size, void* d_ws, size_t ws_size,
                              hipStream_t stream) {
  const float* x1      = (const float*)d_in[0];
  const float* x2      = (const float*)d_in[1];
  const float* norm1_g = (const float*)d_in[2];
  const float* norm1_b = (const float*)d_in[3];
  const float* q_w     = (const float*)d_in[4];
  const float* q_b     = (const float*)d_in[5];
  const float* kv_w    = (const float*)d_in[6];
  const float* kv_b    = (const float*)d_in[7];
  const float* proj_w  = (const float*)d_in[8];
  const float* proj_b  = (const float*)d_in[9];
  const float* rel_t   = (const float*)d_in[10];
  const float* norm3_g = (const float*)d_in[11];
  const float* norm3_b = (const float*)d_in[12];
  const float* fc1_w   = (const float*)d_in[13];
  const float* fc1_b   = (const float*)d_in[14];
  const float* fc2_w   = (const float*)d_in[15];
  const float* fc2_b   = (const float*)d_in[16];

  char* base = (char*)d_ws;
  size_t off = 0;
  auto alloc = [&](size_t bytes) -> char* {
    char* p = base + off;
    off += (bytes + 255) & ~(size_t)255;
    return p;
  };
  _Float16* x1w  = (_Float16*)alloc((size_t)ROWS * CC * 2);        // LN1(x1), windowed
  _Float16* x2w  = (_Float16*)alloc((size_t)ROWS * CC * 2);        // LN1(x2), windowed
  _Float16* qb   = (_Float16*)alloc((size_t)ROWS * CC * 2);        // per-head Q
  _Float16* kb   = (_Float16*)alloc((size_t)ROWS * CC * 2);        // per-head K
  _Float16* vb   = (_Float16*)alloc((size_t)ROWS * CC * 2);        // per-head V
  _Float16* aout = (_Float16*)alloc((size_t)ROWS * CC * 2);        // attn out (windowed)
  float*    xbuf = (float*)   alloc((size_t)ROWS * CC * 4);        // shortcut + attn
  _Float16* xn   = (_Float16*)alloc((size_t)ROWS * CC * 2);        // LN3(x)
  _Float16* h1   = (_Float16*)alloc((size_t)ROWS * 4 * CC * 2);    // gelu(fc1)
  _Float16* qwt  = (_Float16*)alloc((size_t)CC * CC * 2);
  _Float16* kvwt = (_Float16*)alloc((size_t)CC * 2 * CC * 2);
  _Float16* pwt  = (_Float16*)alloc((size_t)CC * CC * 2);
  _Float16* f1wt = (_Float16*)alloc((size_t)CC * 4 * CC * 2);
  _Float16* f2wt = (_Float16*)alloc((size_t)4 * CC * CC * 2);
  (void)ws_size; (void)in_sizes; (void)n_in; (void)out_size;

  // 1) transpose + f16-convert weights (K x N -> N x K)
  auto wp = [&](const float* W, _Float16* Wt, int K, int Nn) {
    int n = K * Nn;
    wprep_kernel<<<(n + 255) / 256, 256, 0, stream>>>(W, Wt, K, Nn);
  };
  wp(q_w,   qwt,  CC, CC);
  wp(kv_w,  kvwt, CC, 2 * CC);
  wp(proj_w, pwt, CC, CC);
  wp(fc1_w, f1wt, CC, 4 * CC);
  wp(fc2_w, f2wt, 4 * CC, CC);

  // 2) LN1 + window partition for both streams
  ln_kernel<<<ROWS, 64, 0, stream>>>(x1, norm1_g, norm1_b, x1w, 1);
  ln_kernel<<<ROWS, 64, 0, stream>>>(x2, norm1_g, norm1_b, x2w, 1);

  // 3) Q = x1w @ q_w + q_b  (scatter into per-head layout)
  gemm_wmma<0, 3><<<dim3(ROWS / 128, CC / 64), 256, 0, stream>>>(
      x1w, qwt, q_b, qb, nullptr, nullptr, ROWS, CC, CC);
  // 4) KV = x2w @ kv_w + kv_b  (split K / V, per-head layout)
  gemm_wmma<0, 4><<<dim3(ROWS / 128, (2 * CC) / 64), 256, 0, stream>>>(
      x2w, kvwt, kv_b, kb, vb, nullptr, ROWS, 2 * CC, CC);

  // 5) windowed multi-head attention: one block per (window, head)
  attn_kernel<<<(ROWS / NTK) * NHD, 128, 0, stream>>>(qb, kb, vb, rel_t, aout);

  // 6) proj + window reverse + residual:  x = x1 + reverse(aout @ proj_w + b)
  gemm_wmma<0, 2><<<dim3(ROWS / 128, CC / 64), 256, 0, stream>>>(
      aout, pwt, proj_b, xbuf, nullptr, x1, ROWS, CC, CC);

  // 7) LN3
  ln_kernel<<<ROWS, 64, 0, stream>>>(xbuf, norm3_g, norm3_b, xn, 0);

  // 8) h1 = gelu(xn @ fc1_w + b)
  gemm_wmma<1, 0><<<dim3(ROWS / 128, (4 * CC) / 64), 256, 0, stream>>>(
      xn, f1wt, fc1_b, h1, nullptr, nullptr, ROWS, 4 * CC, CC);

  // 9) out = x + (h1 @ fc2_w + b)
  gemm_wmma<0, 1><<<dim3(ROWS / 128, CC / 64), 256, 0, stream>>>(
      h1, f2wt, fc2_b, (float*)d_out, nullptr, xbuf, ROWS, CC, 4 * CC);
}
